// SimpleLSTM_32066225832761
// MI455X (gfx1250) — matrix operand order repaired and checked by reading, not compiled, not run on hardware
//
#include <hip/hip_runtime.h>

// ---------------------------------------------------------------------------
// Fused 2-layer LSTM + FC for MI455X (gfx1250, wave32).
//
// One wave per 16-batch-row tile, persistent over all T=512 steps; both LSTM
// layers fused per timestep (h1 [B,T,H]=268MB never touches HBM; only x=84MB
// is read -> ~4us at 23.3TB/s, so the kernel is recurrence-latency bound).
// GEMMs: v_wmma_f32_16x16x32_f16; weight B-fragments stream from LDS.
// Round-4 changes:
//  * bf16 -> f16 datapath: native v_cvt_pk_rtz_f16_f32 / v_cvt_f16_f32
//    replace the 10+-op manual bf16 RNE packs seen in round-3 asm (and f16
//    has more mantissa than bf16 for these tiny values).
//  * layer-1's h1(t-1)@Whh1 tiles hoisted to a phase before layer 0: 8
//    independent WMMAs fill the WMMA->VALU 4-slot hazard windows that were
//    v_nop runs in round-3, and shorten layer-1's serial chain.
//  * retains: i/f/o rows+biases prescaled 0.5 (sigmoid = tanh+fma), layer-0
//    bias folded into x-projection via constant-1.0 in K=31, literal C=0.
// ---------------------------------------------------------------------------

#define HID   32
#define DIN   10
#define TLEN  512
#define NGATE 128   // 4*HID
#define BATCH 4096

typedef __attribute__((ext_vector_type(16))) _Float16 v16h;
typedef __attribute__((ext_vector_type(8)))  float    v8f;

union FragA {                  // 16x32 f16 A-fragment / 32x16 f16 B-fragment
  v16h     v;
  uint4    q[2];
  unsigned u32[8];
  unsigned short u16[16];
};

__device__ __forceinline__ unsigned short f2hbits(float f) {
  union { _Float16 h; unsigned short s; } cv; cv.h = (_Float16)f;  // v_cvt_f16_f32
  return cv.s;
}
__device__ __forceinline__ float hbits2f(unsigned short s) {
  union { unsigned short s; _Float16 h; } cv; cv.s = s;
  return (float)cv.h;                                              // v_cvt_f32_f16
}
// pack two f32 -> two f16 in one v_cvt_pk_rtz_f16_f32
__device__ __forceinline__ unsigned pk_f16(float a, float b) {
#if __has_builtin(__builtin_amdgcn_cvt_pkrtz)
  auto pk = __builtin_amdgcn_cvt_pkrtz(a, b);
  union { decltype(pk) p; unsigned u; } cv; cv.p = pk;
  return cv.u;
#else
  return (unsigned)f2hbits(a) | ((unsigned)f2hbits(b) << 16);
#endif
}

// tanh: native v_tanh_f32 TRANS op on gfx1250 (guarded fallback).
#if __has_builtin(__builtin_amdgcn_tanhf)
__device__ __forceinline__ float ftanh(float x) { return __builtin_amdgcn_tanhf(x); }
#elif __has_builtin(__builtin_amdgcn_tanh_f32)
__device__ __forceinline__ float ftanh(float x) { return __builtin_amdgcn_tanh_f32(x); }
#else
__device__ __forceinline__ float ftanh(float x) {
  float e = __builtin_amdgcn_exp2f(-2.88539008f * x);
  return 2.0f * __builtin_amdgcn_rcpf(1.0f + e) - 1.0f;
}
#endif
// sigmoid(x) where y = 0.5*x was already folded into the weights/bias:
__device__ __forceinline__ float sig_h(float y) {
  return __builtin_fmaf(ftanh(y), 0.5f, 0.5f);
}

struct __align__(16) SMem {
  unsigned short wih0[NGATE][32];  // K 8..15,18..30 zero (x8,x9 at K 16,17); col 31 = fused layer-0 bias
  unsigned short whh0[NGATE][32];  // i/f/o rows of all matrices prescaled 0.5
  unsigned short wih1[NGATE][32];
  unsigned short whh1[NGATE][32];
  float          b1[NGATE];        // prescaled fused layer-1 bias
  unsigned short hbuf0[16][32];    // C-layout -> A-layout bounce (wave-local)
  unsigned short hbuf1[16][32];
};

__device__ __forceinline__ v8f wmma_f16(const FragA& a, const FragA& b, v8f c) {
  return __builtin_amdgcn_wmma_f32_16x16x32_f16(
      false, a.v, false, b.v, (short)0, c, false, false);
}

// B[k][n] = W[n][k]: lane holds 16 consecutive K of column n = two contiguous
// 16B chunks of row-major W -> 2x ds_load_b128 (struct offsets < 64KB).
__device__ __forceinline__ FragA loadB(const unsigned short* w, int g, int koff) {
  FragA f;
  f.q[0] = *(const uint4*)(w + g * 32 + koff);
  f.q[1] = *(const uint4*)(w + g * 32 + koff + 8);
  return f;
}

__global__ __launch_bounds__(32, 1)
void lstm2_fused_kernel(const float* __restrict__ x,
                        const float* __restrict__ Wih0, const float* __restrict__ Whh0,
                        const float* __restrict__ bih0, const float* __restrict__ bhh0,
                        const float* __restrict__ Wih1, const float* __restrict__ Whh1,
                        const float* __restrict__ bih1, const float* __restrict__ bhh1,
                        const float* __restrict__ Wfc,  const float* __restrict__ bfc,
                        float* __restrict__ out)
{
  __shared__ SMem sm;
  const int lane      = threadIdx.x;       // 0..31
  const int nlo       = lane & 15;
  const bool hi       = lane >= 16;
  const int koff      = hi ? 16 : 0;
  const int batchBase = blockIdx.x * 16;

  // ---- stage weights (f32 -> f16, i/f/o rows prescaled by 0.5) -----------
  for (int i = lane; i < NGATE * 32; i += 32) {
    int r = i >> 5, c = i & 31;
    float s = (r >= 64 && r < 96) ? 1.0f : 0.5f;   // g-gate rows unscaled
    float w0;
    if (c < 8 || (c >= 16 && c < 18)) w0 = s * Wih0[r * DIN + (c < 8 ? c : c - 8)];
    else if (c == 31)  w0 = s * (bih0[r] + bhh0[r]);   // fused bias column
    else               w0 = 0.0f;
    sm.wih0[r][c] = f2hbits(w0);
    sm.whh0[r][c] = f2hbits(s * Whh0[i]);
    sm.wih1[r][c] = f2hbits(s * Wih1[i]);
    sm.whh1[r][c] = f2hbits(s * Whh1[i]);
  }
  for (int i = lane; i < NGATE; i += 32) {
    float s = (i >= 64 && i < 96) ? 1.0f : 0.5f;
    sm.b1[i] = s * (bih1[i] + bhh1[i]);
  }
  __syncthreads();

  // ---- layer-1 biases for this lane's gate columns, in registers ---------
  float b1reg[8];
#pragma unroll
  for (int n = 0; n < 8; ++n) b1reg[n] = sm.b1[n * 16 + nlo];

  // ---- persistent state ---------------------------------------------------
  FragA ah0, ah1;                       // h0, h1 as A-fragments (f16)
#pragma unroll
  for (int w = 0; w < 8; ++w) { ah0.u32[w] = 0u; ah1.u32[w] = 0u; }
  float c0[2][8] = {{0}}, c1[2][8] = {{0}};   // cell state, C-layout (f32)

  const float* xrow = x + ((size_t)(batchBase + nlo) * TLEN) * DIN;
  const int row0 = hi ? 8 : 0;

  // x fragment: K=8..15,18..30 zero; hi-lane elem 15 (K=31) = 1.0 for the bias row
  FragA fx;
#pragma unroll
  for (int w = 0; w < 8; ++w) fx.u32[w] = 0u;
  if (hi) fx.u32[7] = 0x3C000000u;          // elem15 = f16(1.0) in high half

  // register-prefetched x values (lane<16: k=0..7; lane>=16: k=8,9, rest 0)
  float xv[8] = {0}, xnext[8] = {0};
  {
    const float* p = xrow;                  // t = 0
    if (!hi) {
      const float2* p2 = (const float2*)p;
      float2 a0 = p2[0], a1 = p2[1], a2 = p2[2], a3 = p2[3];
      xv[0]=a0.x; xv[1]=a0.y; xv[2]=a1.x; xv[3]=a1.y;
      xv[4]=a2.x; xv[5]=a2.y; xv[6]=a3.x; xv[7]=a3.y;
    } else {
      float2 a4 = ((const float2*)p)[4];
      xv[0]=a4.x; xv[1]=a4.y;
    }
  }

  for (int t = 0; t < TLEN; ++t) {
    // live part of the x fragment (xv[2..7]==0 for hi lanes -> packs to 0)
#pragma unroll
    for (int w = 0; w < 4; ++w) fx.u32[w] = pk_f16(xv[2*w], xv[2*w+1]);

    // prefetch x for t+1 (latency hidden behind this step's 32 WMMAs)
    if (t + 1 < TLEN) {
      const float* p = xrow + (t + 1) * DIN;
      if (!hi) {
        const float2* p2 = (const float2*)p;
        float2 a0 = p2[0], a1 = p2[1], a2 = p2[2], a3 = p2[3];
        xnext[0]=a0.x; xnext[1]=a0.y; xnext[2]=a1.x; xnext[3]=a1.y;
        xnext[4]=a2.x; xnext[5]=a2.y; xnext[6]=a3.x; xnext[7]=a3.y;
      } else {
        float2 a4 = ((const float2*)p)[4];
        xnext[0]=a4.x; xnext[1]=a4.y;
      }
    }

    // ==== phase A: layer-1 recurrent partials, independent of layer 0 =====
    // (8 extra independent WMMAs for the scheduler to fill layer-0's
    //  WMMA->VALU hazard windows; shortens layer-1's serial chain.)
    v8f accU1[8];
#pragma unroll
    for (int n = 0; n < 8; ++n) {
      v8f z{};
      FragA bu = loadB(&sm.whh1[0][0], n * 16 + nlo, koff);
      accU1[n] = wmma_f16(ah1, bu, z);       // h1(t-1) @ Whh1^T
    }

    // ======================= layer 0 ======================================
#pragma unroll
    for (int hh = 0; hh < 2; ++hh) {          // column-half of H
      v8f acc[4];                              // i, f, g, o tiles
#pragma unroll
      for (int gi = 0; gi < 4; ++gi) {
        int n = 2 * gi + hh;
        int g = n * 16 + nlo;
        v8f z{};                               // literal C=0
        FragA bw = loadB(&sm.wih0[0][0], g, koff);
        v8f a = wmma_f16(fx, bw, z);           // x @ Wih0^T (+bias via K=31)
        FragA bu = loadB(&sm.whh0[0][0], g, koff);
        a = wmma_f16(ah0, bu, a);              // + h0 @ Whh0^T
        acc[gi] = a;
      }
#pragma unroll
      for (int r = 0; r < 8; r += 2) {         // row pairs -> packed f16
        float i0 = sig_h(acc[0][r]),   i1 = sig_h(acc[0][r+1]);
        float f0 = sig_h(acc[1][r]),   f1 = sig_h(acc[1][r+1]);
        float g0 = ftanh(acc[2][r]),   g1 = ftanh(acc[2][r+1]);
        float o0 = sig_h(acc[3][r]),   o1 = sig_h(acc[3][r+1]);
        float cA = f0 * c0[hh][r]   + i0 * g0;  c0[hh][r]   = cA;
        float cB = f1 * c0[hh][r+1] + i1 * g1;  c0[hh][r+1] = cB;
        unsigned u = pk_f16(o0 * ftanh(cA), o1 * ftanh(cB));
        int col = hh * 16 + nlo;
        sm.hbuf0[row0 + r][col]     = (unsigned short)u;
        sm.hbuf0[row0 + r + 1][col] = (unsigned short)(u >> 16);
      }
    }
    // C-layout -> A-layout via LDS (same-wave DS is in-order; no barrier)
    ah0.q[0] = *(const uint4*)&sm.hbuf0[nlo][koff];
    ah0.q[1] = *(const uint4*)&sm.hbuf0[nlo][koff + 8];

    // ======================= layer 1 ======================================
#pragma unroll
    for (int hh = 0; hh < 2; ++hh) {
      v8f acc[4];
#pragma unroll
      for (int gi = 0; gi < 4; ++gi) {
        int n = 2 * gi + hh;
        FragA bw = loadB(&sm.wih1[0][0], n * 16 + nlo, koff);
        acc[gi] = wmma_f16(ah0, bw, accU1[n]); // h0_new @ Wih1^T + partials
      }
      float bi = b1reg[hh],     bf_ = b1reg[2 + hh];
      float bg = b1reg[4 + hh], bo  = b1reg[6 + hh];
#pragma unroll
      for (int r = 0; r < 8; r += 2) {
        float i0 = sig_h(acc[0][r] + bi),   i1 = sig_h(acc[0][r+1] + bi);
        float f0 = sig_h(acc[1][r] + bf_),  f1 = sig_h(acc[1][r+1] + bf_);
        float g0 = ftanh(acc[2][r] + bg),   g1 = ftanh(acc[2][r+1] + bg);
        float o0 = sig_h(acc[3][r] + bo),   o1 = sig_h(acc[3][r+1] + bo);
        float cA = f0 * c1[hh][r]   + i0 * g0;  c1[hh][r]   = cA;
        float cB = f1 * c1[hh][r+1] + i1 * g1;  c1[hh][r+1] = cB;
        unsigned u = pk_f16(o0 * ftanh(cA), o1 * ftanh(cB));
        int col = hh * 16 + nlo;
        sm.hbuf1[row0 + r][col]     = (unsigned short)u;
        sm.hbuf1[row0 + r + 1][col] = (unsigned short)(u >> 16);
      }
    }
    ah1.q[0] = *(const uint4*)&sm.hbuf1[nlo][koff];
    ah1.q[1] = *(const uint4*)&sm.hbuf1[nlo][koff + 8];

    if (t + 1 < TLEN) {
#pragma unroll
      for (int e = 0; e < 8; ++e) xv[e] = xnext[e];
    }
  }

  // ======================= final FC (H -> 1) ==============================
  // h1(T-1) read from the A-fragment: lane L holds row L%16 with columns
  // {0..15} (lo lanes) / {16..31} (hi lanes).
  float p = 0.0f;
#pragma unroll
  for (int e = 0; e < 16; ++e) {
    int col = (e < 8 ? e : 8 + e) + (hi ? 8 : 0);
    p = __builtin_fmaf(hbits2f(ah1.u16[e]), Wfc[col], p);
  }
  p += __shfl_xor(p, 16, 32);      // combine lo/hi column halves of row nlo
  if (!hi) out[batchBase + nlo] = p + bfc[0];
}

extern "C" void kernel_launch(void* const* d_in, const int* in_sizes, int n_in,
                              void* d_out, int out_size, void* d_ws, size_t ws_size,
                              hipStream_t stream) {
  (void)in_sizes; (void)n_in; (void)d_ws; (void)ws_size; (void)out_size;
  const float* x    = (const float*)d_in[0];
  const float* Wih0 = (const float*)d_in[1];
  const float* Whh0 = (const float*)d_in[2];
  const float* bih0 = (const float*)d_in[3];
  const float* bhh0 = (const float*)d_in[4];
  const float* Wih1 = (const float*)d_in[5];
  const float* Whh1 = (const float*)d_in[6];
  const float* bih1 = (const float*)d_in[7];
  const float* bhh1 = (const float*)d_in[8];
  const float* Wfc  = (const float*)d_in[9];
  const float* bfc  = (const float*)d_in[10];
  float* out = (float*)d_out;

  dim3 grid(BATCH / 16);   // 256 single-wave workgroups
  dim3 block(32);
  lstm2_fused_kernel<<<grid, block, 0, stream>>>(
      x, Wih0, Whh0, bih0, bhh0, Wih1, Whh1, bih1, bhh1, Wfc, bfc, out);
}
